// MultiTaskModel_89283780149841
// MI455X (gfx1250) — matrix operand order repaired and checked by reading, not compiled, not run on hardware
//
#include <hip/hip_runtime.h>

typedef unsigned short u16;
typedef unsigned int   u32;
typedef __bf16 bf16_t;
typedef bf16_t v16bf __attribute__((ext_vector_type(16)));
typedef float  v8f   __attribute__((ext_vector_type(8)));
typedef u32    u32x4 __attribute__((ext_vector_type(4)));

#define BATCH   8192
#define IN_DIM  384
#define HID     512
#define BOT     10
#define BOTP    16
#define ANN     2316

// fp32 -> bf16 round-to-nearest-even (stored as u16)
__device__ __forceinline__ u16 f2bf(float f) {
  union { float f; u32 u; } c; c.f = f;
  u32 u = c.u;
  return (u16)((u + 0x7FFFu + ((u >> 16) & 1u)) >> 16);
}

union FragAB { v16bf v; u32x4 q[2]; };

// CDNA5 async copy: global -> LDS, 16 bytes per lane, tracked by ASYNCcnt.
__device__ __forceinline__ void async_b128_to_lds(u32 lds_off, const void* gptr) {
  asm volatile("global_load_async_to_lds_b128 %0, %1, off"
               :: "v"(lds_off), "v"(gptr) : "memory");
}
__device__ __forceinline__ void wait_async0() {
  asm volatile("s_wait_asynccnt 0" ::: "memory");
}

// ---------------- prep: conversions / transposes ----------------
__global__ void cvt_x(const float* __restrict__ x, u16* __restrict__ xb, int n) {
  int i = blockIdx.x * blockDim.x + threadIdx.x;
  for (; i < n; i += gridDim.x * blockDim.x) xb[i] = f2bf(x[i]);
}

// W1 [IN_DIM, HID] row-major -> w1t [HID, IN_DIM] col-major bf16
__global__ void cvt_w1t(const float* __restrict__ W1, u16* __restrict__ w1t) {
  int i = blockIdx.x * blockDim.x + threadIdx.x;
  if (i < HID * IN_DIM) {
    int n = i / IN_DIM, k = i % IN_DIM;
    w1t[i] = f2bf(W1[k * HID + n]);
  }
}

// W2 [HID, BOT] -> w2t [BOTP, HID] col-major bf16, rows BOT..BOTP-1 zero
__global__ void cvt_w2t(const float* __restrict__ W2, u16* __restrict__ w2t) {
  int i = blockIdx.x * blockDim.x + threadIdx.x;
  if (i < BOTP * HID) {
    int n = i / HID, k = i % HID;
    w2t[i] = (n < BOT) ? f2bf(W2[k * BOT + n]) : (u16)0;
  }
}

// ---------------- GEMM1: h = relu(x @ W1 + b1), bf16 WMMA ----------------
// block = 256 threads = 8 waves (4 along M, 2 along N); wave tile 32x32 (2x2 WMMA).
// B tile (64 cols x 32 K, shared by all M-waves) staged via async-to-LDS, double buffered.
__global__ __launch_bounds__(256) void gemm1_wmma(const u16* __restrict__ xb,
                                                  const u16* __restrict__ w1t,
                                                  const float* __restrict__ b1,
                                                  u16* __restrict__ hb) {
  __shared__ u16 ldsB[2][64 * 32];   // [buf][col*32 + k], 64 B of K per column

  const int t     = threadIdx.x;
  const int lane  = t & 31;
  const int wave  = t >> 5;
  const int wm    = wave & 3;
  const int wn    = wave >> 2;
  const int lrow  = lane & 15;
  const int khalf = lane >> 4;
  const int mBase = blockIdx.x * 128 + wm * 32;
  const int nBlk  = blockIdx.y * 64;

  // async staging: each thread copies one 16B chunk of the B tile
  const int scol   = t >> 2;                 // 0..63 (column within block N-tile)
  const int schunk = t & 3;                  // 0..3  (8 u16 per chunk)
  const u16* gB = w1t + (size_t)(nBlk + scol) * IN_DIM + schunk * 8;

  v8f acc[2][2] = {};
  const u16* aptr[2] = {
    xb + (size_t)(mBase +  0 + lrow) * IN_DIM + khalf * 8,
    xb + (size_t)(mBase + 16 + lrow) * IN_DIM + khalf * 8
  };

  // prologue: stage K-tile 0 into buffer 0
  async_b128_to_lds((u32)(size_t)&ldsB[0][scol * 32 + schunk * 8], gB);

  for (int i = 0; i < IN_DIM / 32; ++i) {
    const int k0  = i * 32;
    const int buf = i & 1;

    wait_async0();        // this thread's staged chunk has landed
    __syncthreads();      // everyone's chunks have landed; prior reads of buf^1 done

    if (k0 + 32 < IN_DIM) // overlap next tile's copy with this tile's WMMAs
      async_b128_to_lds((u32)(size_t)&ldsB[buf ^ 1][scol * 32 + schunk * 8],
                        gB + (k0 + 32));

    FragAB a[2], b[2];
#pragma unroll
    for (int mi = 0; mi < 2; ++mi) {
      a[mi].q[0] = *(const u32x4*)(aptr[mi] + k0);        // K = khalf*8 + 0..7
      a[mi].q[1] = *(const u32x4*)(aptr[mi] + k0 + 16);   // K = 16 + khalf*8 + 0..7
    }
    if (k0 + 32 < IN_DIM) {
      __builtin_prefetch(aptr[0] + k0 + 32, 0, 3);        // near-scope prefetch
      __builtin_prefetch(aptr[1] + k0 + 32, 0, 3);
    }
#pragma unroll
    for (int ni = 0; ni < 2; ++ni) {
      const u16* bp = &ldsB[buf][(wn * 32 + ni * 16 + lrow) * 32 + khalf * 8];
      b[ni].q[0] = *(const u32x4*)(bp);                   // K = khalf*8 + 0..7
      b[ni].q[1] = *(const u32x4*)(bp + 16);              // K = 16 + khalf*8 + 0..7
    }
#pragma unroll
    for (int mi = 0; mi < 2; ++mi)
#pragma unroll
      for (int ni = 0; ni < 2; ++ni)
        acc[mi][ni] = __builtin_amdgcn_wmma_f32_16x16x32_bf16(
            false, a[mi].v, false, b[ni].v, (short)0, acc[mi][ni], false, false);
  }

  // epilogue: + bias, ReLU, convert to bf16 row-major h
#pragma unroll
  for (int mi = 0; mi < 2; ++mi) {
#pragma unroll
    for (int ni = 0; ni < 2; ++ni) {
      const int col  = nBlk + wn * 32 + ni * 16 + lrow;
      const float bs = b1[col];
#pragma unroll
      for (int r = 0; r < 8; ++r) {
        const int row = mBase + mi * 16 + khalf * 8 + r;
        float v = acc[mi][ni][r] + bs;
        v = v > 0.f ? v : 0.f;
        hb[(size_t)row * HID + col] = f2bf(v);
      }
    }
  }
}

// ---------------- GEMM2: f = relu(h @ W2 + b2), one 16x16 tile per wave ----------------
__global__ __launch_bounds__(256) void gemm2_wmma(const u16* __restrict__ hb,
                                                  const u16* __restrict__ w2t,
                                                  const float* __restrict__ b2,
                                                  float* __restrict__ fbuf) {
  const int lane  = threadIdx.x & 31;
  const int wave  = threadIdx.x >> 5;
  const int lrow  = lane & 15;
  const int khalf = lane >> 4;
  const int tile  = blockIdx.x * 8 + wave;   // 0..511
  const int mBase = tile * 16;

  v8f acc = {};
  const u16* ap = hb  + (size_t)(mBase + lrow) * HID + khalf * 8;
  const u16* bp = w2t + (size_t)lrow * HID + khalf * 8;

  for (int k0 = 0; k0 < HID; k0 += 32) {
    FragAB a, b;
    a.q[0] = *(const u32x4*)(ap + k0);
    a.q[1] = *(const u32x4*)(ap + k0 + 16);
    b.q[0] = *(const u32x4*)(bp + k0);
    b.q[1] = *(const u32x4*)(bp + k0 + 16);
    acc = __builtin_amdgcn_wmma_f32_16x16x32_bf16(
        false, a.v, false, b.v, (short)0, acc, false, false);
  }

  const int col  = lrow;
  const float bs = (col < BOT) ? b2[col] : 0.f;
#pragma unroll
  for (int r = 0; r < 8; ++r) {
    const int row = mBase + khalf * 8 + r;
    float v = acc[r] + bs;
    fbuf[(size_t)row * BOTP + col] = v > 0.f ? v : 0.f;
  }
}

// ---------------- heads: out = (f @ Wh + bh) * mask, streaming ----------------
__global__ __launch_bounds__(256) void head_masked(const float* __restrict__ fbuf,
                                                   const float* __restrict__ Wh,
                                                   const float* __restrict__ bh,
                                                   const float* __restrict__ mask,
                                                   float* __restrict__ out) {
  __shared__ float fs[16][BOT];
  const int row0 = blockIdx.y * 16;
  const int t = threadIdx.x;
  if (t < 16 * BOT) {
    const int r = t / BOT, k = t % BOT;
    fs[r][k] = fbuf[(size_t)(row0 + r) * BOTP + k];
  }
  __syncthreads();

  const int col = blockIdx.x * 256 + t;
  if (col >= ANN) return;

  float w[BOT];
#pragma unroll
  for (int k = 0; k < BOT; ++k) w[k] = Wh[(size_t)k * ANN + col];
  const float bias = bh[col];

#pragma unroll 4
  for (int r = 0; r < 16; ++r) {
    const int row = row0 + r;
    float acc = bias;
#pragma unroll
    for (int k = 0; k < BOT; ++k) acc = fmaf(fs[r][k], w[k], acc);
    out[(size_t)row * ANN + col] = acc * mask[(size_t)row * ANN + col];
  }
}

// ---------------- launcher ----------------
extern "C" void kernel_launch(void* const* d_in, const int* in_sizes, int n_in,
                              void* d_out, int out_size, void* d_ws, size_t ws_size,
                              hipStream_t stream) {
  const float* x    = (const float*)d_in[0];
  const float* mask = (const float*)d_in[1];
  const float* W1   = (const float*)d_in[2];
  const float* b1   = (const float*)d_in[3];
  const float* W2   = (const float*)d_in[4];
  const float* b2   = (const float*)d_in[5];
  const float* Wh   = (const float*)d_in[6];
  const float* bh   = (const float*)d_in[7];
  float* out = (float*)d_out;

  char* ws = (char*)d_ws;
  size_t off = 0;
  u16* xb  = (u16*)(ws + off); off += (size_t)BATCH * IN_DIM * sizeof(u16); // 6.29 MB
  u16* w1t = (u16*)(ws + off); off += (size_t)HID * IN_DIM * sizeof(u16);   // 0.39 MB
  u16* w2t = (u16*)(ws + off); off += (size_t)BOTP * HID * sizeof(u16);     // 16 KB
  u16* hb  = (u16*)(ws + off); off += (size_t)BATCH * HID * sizeof(u16);    // 8.39 MB
  float* fbuf = (float*)(ws + off);                                         // 0.52 MB

  cvt_x<<<1024, 256, 0, stream>>>(x, xb, BATCH * IN_DIM);
  cvt_w1t<<<(HID * IN_DIM + 255) / 256, 256, 0, stream>>>(W1, w1t);
  cvt_w2t<<<(BOTP * HID + 255) / 256, 256, 0, stream>>>(W2, w2t);

  gemm1_wmma<<<dim3(BATCH / 128, HID / 64), 256, 0, stream>>>(xb, w1t, b1, hb);
  gemm2_wmma<<<BATCH / (16 * 8), 256, 0, stream>>>(hb, w2t, b2, fbuf);
  head_masked<<<dim3((ANN + 255) / 256, BATCH / 16), 256, 0, stream>>>(fbuf, Wh, bh, mask, out);
}
